// ConvAttention_55851754717529
// MI455X (gfx1250) — compile-verified, matrix operand order checked by reading, not run on hardware
//
#include <hip/hip_runtime.h>
#include <hip/hip_bf16.h>

// ---------------------------------------------------------------------------
// CDNA5 (gfx1250) implementation of the ConvAttention block.
// Precision: bf16 WMMA (v_wmma_f32_16x16x32_bf16) with f32 accumulation for
// all GEMMs; f32 VALU for softmax / GELU / depthwise conv.
// Context GEMM uses the Tensor Data Mover (tensor_load_to_lds) with
// double-buffered LDS staging tracked via TENSORcnt.
// ---------------------------------------------------------------------------

typedef __bf16 v16bf __attribute__((ext_vector_type(16)));
typedef __bf16 v8bf  __attribute__((ext_vector_type(8)));
typedef float  v8f   __attribute__((ext_vector_type(8)));
typedef unsigned int u32x4 __attribute__((ext_vector_type(4)));
typedef int i32x4 __attribute__((ext_vector_type(4)));
typedef int i32x8 __attribute__((ext_vector_type(8)));

struct __align__(16) Bf16x8 { __hip_bfloat16 h[8]; };

#define B_SZ     16
#define C_SZ     256
#define HW_SZ    4096
#define HEADS    8
#define HIDDEN   64
#define TMP_CH   512          // HEADS*HIDDEN
#define BH_SZ    (B_SZ * HEADS)

#if __has_builtin(__builtin_amdgcn_tensor_load_to_lds) && \
    __has_builtin(__builtin_amdgcn_s_wait_tensorcnt)
#define HAVE_TDM 1
#else
#define HAVE_TDM 0
#endif

// Build a 16x32 bf16 A/B WMMA operand from two 16-byte chunks:
// lane L: row/col = L&15, K-low-half = (L>>4)*8, plus the +16 half.
__device__ __forceinline__ v16bf make_ab(const void* plo, const void* phi) {
    v8bf lo = *reinterpret_cast<const v8bf*>(plo);
    v8bf hi = *reinterpret_cast<const v8bf*>(phi);
    return __builtin_shufflevector(lo, hi,
        0, 1, 2, 3, 4, 5, 6, 7, 8, 9, 10, 11, 12, 13, 14, 15);
}

#if HAVE_TDM
// Issue a TDM 2D tile load: global (row-major, stride_d0 elements/row,
// 2-byte elements) -> LDS at byte offset lds_off.  g1w0 carries
// {pad_amount,pad_interval,pad_enable,data_size} per D# group-1 word 0.
__device__ __forceinline__ void tdm_load_2d(
    const void* gaddr, unsigned lds_off,
    unsigned tile_d0, unsigned tile_d1,
    unsigned tensor_d0, unsigned tensor_d1,
    unsigned stride_d0, unsigned g1w0)
{
    const unsigned long long ga = (unsigned long long)gaddr;
    u32x4 g0;
    g0[0] = 1u;                                       // count = 1 (valid D#)
    g0[1] = lds_off;                                  // LDS byte address
    g0[2] = (unsigned)(ga & 0xffffffffu);             // global_addr[31:0]
    g0[3] = (unsigned)((ga >> 32) & 0x01ffffffu) |    // global_addr[56:32]
            (2u << 30);                               // type = 2 ("image")
    i32x8 g1;
    g1[0] = (int)g1w0;                                // mask=0, data_size, pad
    g1[1] = (int)(tensor_d0 << 16);                   // tensor_dim0 @ bit48
    g1[2] = (int)((tensor_d0 >> 16) | (tensor_d1 << 16));
    g1[3] = (int)((tensor_d1 >> 16) | (tile_d0 << 16));
    g1[4] = (int)(tile_d1 & 0xffffu);                 // tile_dim1, tile_dim2=0
    g1[5] = (int)stride_d0;                           // dim0 stride low 32
    g1[6] = 0;                                        // stride hi, dim1 stride
    g1[7] = 0;
    const i32x4 gz = {0, 0, 0, 0};
#if __clang_major__ >= 23
    const i32x8 gz8 = {0, 0, 0, 0, 0, 0, 0, 0};
    __builtin_amdgcn_tensor_load_to_lds(g0, g1, gz, gz, gz8, 0);
#else
    __builtin_amdgcn_tensor_load_to_lds(g0, g1, gz, gz, 0);
#endif
}
#endif  // HAVE_TDM

// ---------------------------------------------------------------------------
// 0) f32 -> bf16 weight conversion
// ---------------------------------------------------------------------------
__global__ void f32_to_bf16_kernel(const float* __restrict__ src,
                                   __hip_bfloat16* __restrict__ dst, int n) {
    int i = blockIdx.x * 256 + threadIdx.x;
    if (i < n) dst[i] = __float2bfloat16(src[i]);
}

// ---------------------------------------------------------------------------
// 1) depthwise 3x3 conv + transpose both fmap and dw to [b][p][c] bf16
// ---------------------------------------------------------------------------
__global__ __launch_bounds__(256) void prep_dw_kernel(
    const float* __restrict__ fmap,   // [16][256][64][64]
    const float* __restrict__ Wdw,    // [256][1][3][3]
    __hip_bfloat16* __restrict__ fmapT,  // [16][4096][256]
    __hip_bfloat16* __restrict__ dwT)    // [16][4096][256]
{
    const int p = blockIdx.x * 32 + threadIdx.x;   // pixel
    const int c = blockIdx.y * 8 + threadIdx.y;    // channel
    const int b = blockIdx.z;
    const int h = p >> 6, w = p & 63;

    const float* fc = fmap + ((size_t)b * C_SZ + c) * HW_SZ;
    const float* wk = Wdw + c * 9;

    float acc = 0.0f;
    #pragma unroll
    for (int i = 0; i < 3; ++i) {
        const int hh = h + i - 1;
        if (hh < 0 || hh >= 64) continue;
        #pragma unroll
        for (int j = 0; j < 3; ++j) {
            const int ww = w + j - 1;
            if (ww < 0 || ww >= 64) continue;
            acc += wk[i * 3 + j] * fc[hh * 64 + ww];
        }
    }
    const size_t tidx = ((size_t)b * HW_SZ + p) * C_SZ + c;
    fmapT[tidx] = __float2bfloat16(fc[p]);
    dwT[tidx]   = __float2bfloat16(acc);
}

// ---------------------------------------------------------------------------
// 2) q projection + softmax over feature dim + scale; stored q_sm[bh][n][64]
// ---------------------------------------------------------------------------
__global__ __launch_bounds__(128) void proj_q_kernel(
    const __hip_bfloat16* __restrict__ Wq_bf,   // [512][256]
    const __hip_bfloat16* __restrict__ fmapT,   // [16][4096][256]
    __hip_bfloat16* __restrict__ q_sm)          // [128][4096][64]
{
    const int lane = threadIdx.x & 31;
    const int wave = threadIdx.x >> 5;
    const int lr   = lane & 15;
    const int klo  = (lane >> 4) * 8;
    const int b    = blockIdx.z;
    const int head = blockIdx.y;
    const int nbase = (blockIdx.x * 4 + wave) * 16;

    v8f acc[4];
    #pragma unroll
    for (int t = 0; t < 4; ++t) acc[t] = {};

    const __hip_bfloat16* xcol =
        fmapT + ((size_t)b * HW_SZ + nbase + lr) * C_SZ;

    for (int kk = 0; kk < C_SZ; kk += 32) {
        v16bf bm = make_ab(xcol + kk + klo, xcol + kk + klo + 16);
        #pragma unroll
        for (int t = 0; t < 4; ++t) {
            const __hip_bfloat16* arow =
                Wq_bf + (size_t)(head * HIDDEN + t * 16 + lr) * C_SZ + kk + klo;
            v16bf am = make_ab(arow, arow + 16);
            acc[t] = __builtin_amdgcn_wmma_f32_16x16x32_bf16(
                false, am, false, bm, (short)0, acc[t], false, false);
        }
    }

    // Softmax over d (64): lane l and l^16 hold complementary halves of d
    // for the same token (N = lane&15).
    float m = -1e30f;
    #pragma unroll
    for (int t = 0; t < 4; ++t)
        #pragma unroll
        for (int r = 0; r < 8; ++r) m = fmaxf(m, acc[t][r]);
    m = fmaxf(m, __shfl_xor(m, 16, 32));

    float ex[4][8];
    float s = 0.0f;
    #pragma unroll
    for (int t = 0; t < 4; ++t)
        #pragma unroll
        for (int r = 0; r < 8; ++r) {
            ex[t][r] = __expf(acc[t][r] - m);
            s += ex[t][r];
        }
    s += __shfl_xor(s, 16, 32);
    const float inv = 0.125f / s;   // SCALE = HIDDEN^-0.5 = 1/8

    const int n = nbase + lr;
    __hip_bfloat16* qrow =
        q_sm + (((size_t)b * HEADS + head) * HW_SZ + n) * HIDDEN;
    #pragma unroll
    for (int t = 0; t < 4; ++t) {
        Bf16x8 pk;
        #pragma unroll
        for (int r = 0; r < 8; ++r) pk.h[r] = __float2bfloat16(ex[t][r] * inv);
        *reinterpret_cast<Bf16x8*>(qrow + t * 16 + klo) = pk;  // d contiguous
    }
}

// ---------------------------------------------------------------------------
// 3) k/v projection (Wkv x dw), stored channel-major [bh][64][4096] bf16.
// ---------------------------------------------------------------------------
__global__ __launch_bounds__(128) void proj_kv_kernel(
    const __hip_bfloat16* __restrict__ Wkv_bf,  // [1024][256]
    const __hip_bfloat16* __restrict__ dwT,     // [16][4096][256]
    __hip_bfloat16* __restrict__ kbuf,          // [128][64][4096]
    __hip_bfloat16* __restrict__ vbuf)          // [128][64][4096]
{
    const int lane = threadIdx.x & 31;
    const int wave = threadIdx.x >> 5;
    const int lr   = lane & 15;
    const int hl   = lane >> 4;
    const int klo  = hl * 8;
    const int b     = blockIdx.z;
    const int obase = blockIdx.y * 64;           // 16 M-blocks over 1024 ch
    const int nbase = (blockIdx.x * 4 + wave) * 16;

    v8f acc[4];
    #pragma unroll
    for (int t = 0; t < 4; ++t) acc[t] = {};

    const __hip_bfloat16* xcol =
        dwT + ((size_t)b * HW_SZ + nbase + lr) * C_SZ;

    for (int kk = 0; kk < C_SZ; kk += 32) {
        v16bf bm = make_ab(xcol + kk + klo, xcol + kk + klo + 16);
        #pragma unroll
        for (int t = 0; t < 4; ++t) {
            const __hip_bfloat16* arow =
                Wkv_bf + (size_t)(obase + t * 16 + lr) * C_SZ + kk + klo;
            v16bf am = make_ab(arow, arow + 16);
            acc[t] = __builtin_amdgcn_wmma_f32_16x16x32_bf16(
                false, am, false, bm, (short)0, acc[t], false, false);
        }
    }

    const int p = nbase + lr;
    #pragma unroll
    for (int t = 0; t < 4; ++t)
        #pragma unroll
        for (int r = 0; r < 8; ++r) {
            const int o = obase + t * 16 + r + 8 * hl;
            const int oc = o & 511;
            const int head = oc >> 6, d = oc & 63;
            __hip_bfloat16* dst = (o < TMP_CH) ? kbuf : vbuf;
            dst[(((size_t)b * HEADS + head) * HIDDEN + d) * HW_SZ + p] =
                __float2bfloat16(acc[t][r]);
        }
}

// ---------------------------------------------------------------------------
// 4) k softmax over the token dim (4096) per (bh, d) row, in place.
// ---------------------------------------------------------------------------
__global__ __launch_bounds__(256) void k_softmax_kernel(
    __hip_bfloat16* __restrict__ kbuf)          // [8192][4096]
{
    __shared__ float red[256];
    const int tid = threadIdx.x;
    __hip_bfloat16* kr = kbuf + (size_t)blockIdx.x * HW_SZ;

    float v[16];
    float m = -1e30f;
    #pragma unroll
    for (int j = 0; j < 16; ++j) {
        v[j] = __bfloat162float(kr[tid + j * 256]);
        m = fmaxf(m, v[j]);
    }
    red[tid] = m; __syncthreads();
    for (int s = 128; s > 0; s >>= 1) {
        if (tid < s) red[tid] = fmaxf(red[tid], red[tid + s]);
        __syncthreads();
    }
    m = red[0]; __syncthreads();

    float sum = 0.0f;
    #pragma unroll
    for (int j = 0; j < 16; ++j) { v[j] = __expf(v[j] - m); sum += v[j]; }
    red[tid] = sum; __syncthreads();
    for (int s = 128; s > 0; s >>= 1) {
        if (tid < s) red[tid] += red[tid + s];
        __syncthreads();
    }
    const float inv = 1.0f / red[0];
    #pragma unroll
    for (int j = 0; j < 16; ++j)
        kr[tid + j * 256] = __float2bfloat16(v[j] * inv);
}

// ---------------------------------------------------------------------------
// 5) context = k_sm^T v : [bh] 64x64, K = 4096.  Stored ctxT[bh][e][d] bf16.
//    TDM version: wave0 DMAs k panel chunks, wave1 DMAs v panel chunks into
//    double-buffered LDS (padded rows via the TDM pad feature), tracked with
//    TENSORcnt; all 4 waves compute from LDS with ds_load_b128.
// ---------------------------------------------------------------------------
#define CTX_CK   64            // K-chunk per TDM transfer
#define CTX_PAD  8             // 8 bf16 = 4 DWORDs pad per 32-DWORD row
#define CTX_ROW  (CTX_CK + CTX_PAD)

__global__ __launch_bounds__(128) void context_kernel(
    const __hip_bfloat16* __restrict__ kbuf,    // [128][64][4096]
    const __hip_bfloat16* __restrict__ vbuf,    // [128][64][4096]
    __hip_bfloat16* __restrict__ ctxT)          // [128][64][64]
{
    const int lane = threadIdx.x & 31;
    const int wave = threadIdx.x >> 5;          // e-tile
    const int lr   = lane & 15;
    const int hl   = lane >> 4;
    const int klo  = hl * 8;
    const int bh   = blockIdx.x;

    v8f acc[4];
    #pragma unroll
    for (int t = 0; t < 4; ++t) acc[t] = {};

    const __hip_bfloat16* kbase = kbuf + (size_t)bh * HIDDEN * HW_SZ;
    const __hip_bfloat16* vbase = vbuf + (size_t)bh * HIDDEN * HW_SZ;

#if HAVE_TDM
    // double-buffered padded tiles: 2 * 64 * 72 * 2B = 18 KB each, 36 KB total
    __shared__ __hip_bfloat16 kTile[2][HIDDEN][CTX_ROW];
    __shared__ __hip_bfloat16 vTile[2][HIDDEN][CTX_ROW];

    // D# group1 word0: data_size=2B (1<<16), pad_enable (1<<20),
    // pad_interval=32 DWORDs (4<<22), pad_amount=4 DWORDs (3<<25).
    const unsigned G1W0 = (1u << 16) | (1u << 20) | (4u << 22) | (3u << 25);
    const int nchunks = HW_SZ / CTX_CK;         // 64

    // prologue: chunk 0 -> buffer 0  (TDM ignores EXEC; wave-uniform branch)
    if (wave == 0)
        tdm_load_2d(kbase, (unsigned)(size_t)&kTile[0][0][0],
                    CTX_CK, HIDDEN, HW_SZ, HIDDEN, HW_SZ, G1W0);
    if (wave == 1)
        tdm_load_2d(vbase, (unsigned)(size_t)&vTile[0][0][0],
                    CTX_CK, HIDDEN, HW_SZ, HIDDEN, HW_SZ, G1W0);

    for (int ck = 0; ck < nchunks; ++ck) {
        const int buf = ck & 1;
        const bool more = (ck + 1) < nchunks;
        if (more) {
            // previous compute on buf^1 finished at last-iteration barrier
            if (wave == 0)
                tdm_load_2d(kbase + (size_t)(ck + 1) * CTX_CK,
                            (unsigned)(size_t)&kTile[buf ^ 1][0][0],
                            CTX_CK, HIDDEN, HW_SZ, HIDDEN, HW_SZ, G1W0);
            if (wave == 1)
                tdm_load_2d(vbase + (size_t)(ck + 1) * CTX_CK,
                            (unsigned)(size_t)&vTile[buf ^ 1][0][0],
                            CTX_CK, HIDDEN, HW_SZ, HIDDEN, HW_SZ, G1W0);
        }
        if (wave < 2) { // wait until the chunk for THIS iteration has landed
            if (more) __builtin_amdgcn_s_wait_tensorcnt(1);
            else      __builtin_amdgcn_s_wait_tensorcnt(0);
        }
        __syncthreads();

        #pragma unroll
        for (int ks = 0; ks < CTX_CK; ks += 32) {
            v16bf bm = make_ab(&vTile[buf][wave * 16 + lr][ks + klo],
                               &vTile[buf][wave * 16 + lr][ks + klo + 16]);
            #pragma unroll
            for (int t = 0; t < 4; ++t) {
                v16bf am = make_ab(&kTile[buf][t * 16 + lr][ks + klo],
                                   &kTile[buf][t * 16 + lr][ks + klo + 16]);
                acc[t] = __builtin_amdgcn_wmma_f32_16x16x32_bf16(
                    false, am, false, bm, (short)0, acc[t], false, false);
            }
        }
        __syncthreads();   // buffer may be overwritten next iteration
    }
#else
    const __hip_bfloat16* vcol = vbase + (size_t)(wave * 16 + lr) * HW_SZ;
    for (int kk = 0; kk < HW_SZ; kk += 32) {
        __builtin_prefetch(vcol + kk + 1024, 0, 1);   // global_prefetch_b8
        v16bf bm = make_ab(vcol + kk + klo, vcol + kk + klo + 16);
        #pragma unroll
        for (int t = 0; t < 4; ++t) {
            const __hip_bfloat16* arow =
                kbase + (size_t)(t * 16 + lr) * HW_SZ + kk + klo;
            v16bf am = make_ab(arow, arow + 16);
            acc[t] = __builtin_amdgcn_wmma_f32_16x16x32_bf16(
                false, am, false, bm, (short)0, acc[t], false, false);
        }
    }
#endif

    const int e = wave * 16 + lr;
    __hip_bfloat16* crow = ctxT + ((size_t)bh * HIDDEN + e) * HIDDEN;
    #pragma unroll
    for (int t = 0; t < 4; ++t) {
        Bf16x8 pk;
        #pragma unroll
        for (int r = 0; r < 8; ++r) pk.h[r] = __float2bfloat16(acc[t][r]);
        *reinterpret_cast<Bf16x8*>(crow + t * 16 + klo) = pk;  // d contiguous
    }
}

// ---------------------------------------------------------------------------
// 6) fused: out_attn = ctx^T x q_sm per head -> GELU -> LDS [tok][512]
//    -> Wout GEMM (+bias) -> f32 output [b][256][4096].
// ---------------------------------------------------------------------------
__global__ __launch_bounds__(128) void attn_out_kernel(
    const __hip_bfloat16* __restrict__ q_sm,    // [128][4096][64]
    const __hip_bfloat16* __restrict__ ctxT,    // [128][64][64]
    const __hip_bfloat16* __restrict__ Wout_bf, // [256][512]
    const float* __restrict__ bout,             // [256]
    float* __restrict__ out)                    // [16][256][4096]
{
    __shared__ __hip_bfloat16 g_lds[16][TMP_CH];   // 16 KB

    const int lane = threadIdx.x & 31;
    const int wave = threadIdx.x >> 5;
    const int lr   = lane & 15;
    const int hl   = lane >> 4;
    const int klo  = hl * 8;
    const int b     = blockIdx.y;
    const int nbase = blockIdx.x * 16;

    // ---- GEMM1: g[e, tok] = sum_d ctxT[e][d] * q_sm[tok][d], then GELU ----
    #pragma unroll
    for (int hh = 0; hh < 2; ++hh) {
        const int head = wave + hh * 4;
        const size_t bh = (size_t)b * HEADS + head;

        v8f acc[4];
        #pragma unroll
        for (int t = 0; t < 4; ++t) acc[t] = {};

        const __hip_bfloat16* qcol =
            q_sm + (bh * HW_SZ + nbase + lr) * HIDDEN;
        const __hip_bfloat16* cbase = ctxT + bh * HIDDEN * HIDDEN;

        #pragma unroll
        for (int kk = 0; kk < HIDDEN; kk += 32) {
            v16bf bm = make_ab(qcol + kk + klo, qcol + kk + klo + 16);
            #pragma unroll
            for (int t = 0; t < 4; ++t) {
                const __hip_bfloat16* arow =
                    cbase + (size_t)(t * 16 + lr) * HIDDEN + kk + klo;
                v16bf am = make_ab(arow, arow + 16);
                acc[t] = __builtin_amdgcn_wmma_f32_16x16x32_bf16(
                    false, am, false, bm, (short)0, acc[t], false, false);
            }
        }

        const int tok = lr;
        #pragma unroll
        for (int t = 0; t < 4; ++t) {
            Bf16x8 pk;
            #pragma unroll
            for (int r = 0; r < 8; ++r) {
                const float x = acc[t][r];
                const float g = 0.5f * x * (1.0f + erff(x * 0.70710678f));
                pk.h[r] = __float2bfloat16(g);
            }
            *reinterpret_cast<Bf16x8*>(
                &g_lds[tok][head * HIDDEN + t * 16 + klo]) = pk;
        }
    }
    __syncthreads();

    // ---- GEMM2: y[co, tok] = sum_o Wout[co][o] * g[tok][o] + bout[co] ----
    v8f acc2[4];
    #pragma unroll
    for (int t = 0; t < 4; ++t) acc2[t] = {};

    const __hip_bfloat16* gcol = &g_lds[lr][0];
    #pragma unroll
    for (int kk = 0; kk < TMP_CH; kk += 32) {
        v16bf bm = make_ab(gcol + kk + klo, gcol + kk + klo + 16);
        #pragma unroll
        for (int t = 0; t < 4; ++t) {
            const __hip_bfloat16* arow =
                Wout_bf + (size_t)(wave * 64 + t * 16 + lr) * TMP_CH + kk + klo;
            v16bf am = make_ab(arow, arow + 16);
            acc2[t] = __builtin_amdgcn_wmma_f32_16x16x32_bf16(
                false, am, false, bm, (short)0, acc2[t], false, false);
        }
    }

    const int p = nbase + lr;
    #pragma unroll
    for (int t = 0; t < 4; ++t)
        #pragma unroll
        for (int r = 0; r < 8; ++r) {
            const int co = wave * 64 + t * 16 + r + 8 * hl;
            out[((size_t)b * C_SZ + co) * HW_SZ + p] = acc2[t][r] + bout[co];
        }
}

// ---------------------------------------------------------------------------
// launcher
// ---------------------------------------------------------------------------
extern "C" void kernel_launch(void* const* d_in, const int* in_sizes, int n_in,
                              void* d_out, int out_size, void* d_ws, size_t ws_size,
                              hipStream_t stream) {
    const float* fmap = (const float*)d_in[0];
    const float* Wq   = (const float*)d_in[1];
    const float* Wdw  = (const float*)d_in[2];
    const float* Wkv  = (const float*)d_in[3];
    const float* Wout = (const float*)d_in[4];
    const float* bout = (const float*)d_in[5];
    float* out = (float*)d_out;

    // workspace layout (all regions 16B-aligned by construction)
    char* ws = (char*)d_ws;
    const size_t n_px   = (size_t)B_SZ * HW_SZ;              // 65536
    __hip_bfloat16* fmapT  = (__hip_bfloat16*)ws;            ws += n_px * C_SZ * 2;            // 32 MB
    __hip_bfloat16* dwT    = (__hip_bfloat16*)ws;            ws += n_px * C_SZ * 2;            // 32 MB
    __hip_bfloat16* Wq_bf  = (__hip_bfloat16*)ws;            ws += (size_t)TMP_CH * C_SZ * 2;
    __hip_bfloat16* Wkv_bf = (__hip_bfloat16*)ws;            ws += (size_t)2 * TMP_CH * C_SZ * 2;
    __hip_bfloat16* Wout_bf= (__hip_bfloat16*)ws;            ws += (size_t)C_SZ * TMP_CH * 2;
    __hip_bfloat16* q_sm   = (__hip_bfloat16*)ws;            ws += (size_t)BH_SZ * HW_SZ * HIDDEN * 2;  // 64 MB
    __hip_bfloat16* kbuf   = (__hip_bfloat16*)ws;            ws += (size_t)BH_SZ * HIDDEN * HW_SZ * 2;  // 64 MB
    __hip_bfloat16* vbuf   = (__hip_bfloat16*)ws;            ws += (size_t)BH_SZ * HIDDEN * HW_SZ * 2;  // 64 MB
    __hip_bfloat16* ctxT   = (__hip_bfloat16*)ws;            ws += (size_t)BH_SZ * HIDDEN * HIDDEN * 2; // 1 MB

    // 0) weight conversion
    {
        int nq = TMP_CH * C_SZ;        // 131072
        int nk = 2 * TMP_CH * C_SZ;    // 262144
        int no = C_SZ * TMP_CH;        // 131072
        f32_to_bf16_kernel<<<(nq + 255) / 256, 256, 0, stream>>>(Wq,   Wq_bf,  nq);
        f32_to_bf16_kernel<<<(nk + 255) / 256, 256, 0, stream>>>(Wkv,  Wkv_bf, nk);
        f32_to_bf16_kernel<<<(no + 255) / 256, 256, 0, stream>>>(Wout, Wout_bf, no);
    }
    // 1) depthwise conv + transposed bf16 copies
    prep_dw_kernel<<<dim3(128, 32, 16), dim3(32, 8), 0, stream>>>(fmap, Wdw, fmapT, dwT);
    // 2) q projection + feature softmax
    proj_q_kernel<<<dim3(64, HEADS, B_SZ), 128, 0, stream>>>(Wq_bf, fmapT, q_sm);
    // 3) k/v projection
    proj_kv_kernel<<<dim3(64, 16, B_SZ), 128, 0, stream>>>(Wkv_bf, dwT, kbuf, vbuf);
    // 4) k softmax over tokens
    k_softmax_kernel<<<BH_SZ * HIDDEN, 256, 0, stream>>>(kbuf);
    // 5) context GEMM (TDM double-buffered LDS staging when available)
    context_kernel<<<BH_SZ, 128, 0, stream>>>(kbuf, vbuf, ctxT);
    // 6) fused attention-out + GELU + Wout projection
    attn_out_kernel<<<dim3(HW_SZ / 16, B_SZ), 128, 0, stream>>>(q_sm, ctxT, Wout_bf, bout, out);
}
